// DeltaNetModel_38740605010043
// MI455X (gfx1250) — matrix operand order repaired and checked by reading, not compiled, not run on hardware
//
#include <hip/hip_runtime.h>
#include <hip/hip_bf16.h>

// ---------------- model dims ----------------
#define S_LEN   1024
#define BATCH   32
#define IN_DIM  3072
#define HID     256
#define NHEAD   16
#define HD      16
#define DFF     1024
#define NLAYER  4
#define NCLS    1623
#define EMB_DIM 10
#define NTOK    (S_LEN * BATCH)         // 32768
#define QKVB    (NHEAD * (3 * HD + 1))  // 784

typedef __bf16 bf16x16 __attribute__((ext_vector_type(16)));
typedef float  f32x8   __attribute__((ext_vector_type(8)));
typedef int    v4i     __attribute__((__vector_size__(16)));

// ---------------- bf16 pack: v_cvt_pk_bf16_f32 ----------------
__device__ __forceinline__ unsigned int pack_bf16x2(float a, float b) {
#if __has_builtin(__builtin_amdgcn_cvt_pk_bf16_f32)
    auto p = __builtin_amdgcn_cvt_pk_bf16_f32(a, b);
    return __builtin_bit_cast(unsigned int, p);
#else
    unsigned int r;
    asm("v_cvt_pk_bf16_f32 %0, %1, %2" : "=v"(r) : "v"(a), "v"(b));
    return r;
#endif
}

// ---------------- async global->LDS (CDNA5) with fallback ----------------
__device__ __forceinline__ void async_copy_b128(const float* g, float* l) {
#if __has_builtin(__builtin_amdgcn_global_load_async_to_lds_b128)
    __builtin_amdgcn_global_load_async_to_lds_b128(
        (__attribute__((address_space(1))) v4i*)g,
        (__attribute__((address_space(3))) v4i*)l, 0, 0);
#else
    *(float4*)l = *(const float4*)g;
#endif
}

template <int N>
__device__ __forceinline__ void wait_async() {
#if __has_builtin(__builtin_amdgcn_s_wait_asynccnt)
    __builtin_amdgcn_s_wait_asynccnt(N);
#else
#if __has_builtin(__builtin_amdgcn_global_load_async_to_lds_b128)
    if constexpr (N == 0) asm volatile("s_wait_asynccnt 0x0" ::: "memory");
    else                  asm volatile("s_wait_asynccnt 0x1" ::: "memory");
#endif
#endif
}

#define LDS_PITCH 20   // dwords per 32-element bf16 row (16 data + 4 pad), 16B aligned

// A fragment: 16x32 (MxK) bf16. lane = half*16 + m.
__device__ __forceinline__ bf16x16 load_frag_a(const unsigned int* row, int half) {
    union { unsigned int u[8]; bf16x16 v; } f;
    const uint4* p = (const uint4*)row;
    uint4 lo = p[half];
    uint4 hi = p[2 + half];
    f.u[0] = lo.x; f.u[1] = lo.y; f.u[2] = lo.z; f.u[3] = lo.w;
    f.u[4] = hi.x; f.u[5] = hi.y; f.u[6] = hi.z; f.u[7] = hi.w;
    return f.v;
}

// B fragment: 32x16 (KxN) bf16, stored as weight rows (N rows x K cols).
__device__ __forceinline__ bf16x16 load_frag_b(const unsigned int* row, int half) {
    union { unsigned int u[8]; bf16x16 v; } f;
    const uint4* p = (const uint4*)row;
    uint4 lo = p[half];
    uint4 hi = p[2 + half];
    f.u[0] = lo.x; f.u[1] = lo.y; f.u[2] = lo.z; f.u[3] = lo.w;
    f.u[4] = hi.x; f.u[5] = hi.y; f.u[6] = hi.z; f.u[7] = hi.w;
    return f.v;
}

// ---------------- f32 -> bf16 bulk convert (for weights, once per launch) ----------------
__global__ __launch_bounds__(256)
void cvt_f32_bf16_kernel(const float* __restrict__ src, __bf16* __restrict__ dst, int n2)
{
    int i = blockIdx.x * 256 + threadIdx.x;
    if (i < n2) {
        float2 f = *(const float2*)(src + 2 * (size_t)i);
        *(unsigned int*)(dst + 2 * (size_t)i) = pack_bf16x2(f.x, f.y);
    }
}

// ---------------- WMMA GEMM: C[M,N] = A_f32[M,K] * Wbf16[N,K]^T (+bias)(+res)(relu) ----------------
// grid = (M/128, ceil(N/128)), block = 256 (8 waves; each wave owns a 32x64 sub-tile)
// Double-buffered LDS; next K-tile loaded to registers while WMMAs consume current tile.
__global__ __launch_bounds__(256)
void gemm_wmma_bf16(const float* __restrict__ A, int lda,
                    const __bf16* __restrict__ Bw, int ldb,
                    const float* __restrict__ bias,
                    const float* __restrict__ res,
                    float* __restrict__ C,
                    int N, int K, int relu)
{
    __shared__ __align__(16) unsigned int sA[2][128 * LDS_PITCH];
    __shared__ __align__(16) unsigned int sB[2][128 * LDS_PITCH];

    const int t    = threadIdx.x;
    const int lane = t & 31;
    const int wave = t >> 5;
    const int wm   = wave >> 1;   // 0..3 -> M offset 32*wm
    const int wn   = wave & 1;    // 0..1 -> N offset 64*wn
    const int half = lane >> 4;
    const int nl   = lane & 15;
    const long long bm = (long long)blockIdx.x * 128;
    const int bn = blockIdx.y * 128;

    f32x8 acc[2][4];
#pragma unroll
    for (int mi = 0; mi < 2; ++mi)
#pragma unroll
        for (int ni = 0; ni < 4; ++ni)
#pragma unroll
            for (int r = 0; r < 8; ++r) acc[mi][ni][r] = 0.f;

    float2 ra[8];
    unsigned int rbu[8];
    auto load_global = [&](int k0) {
#pragma unroll
        for (int it = 0; it < 8; ++it) {
            int idx = t + it * 256;                 // 0..2047
            int r = idx >> 4, c = idx & 15;
            ra[it] = *(const float2*)(A + (size_t)(bm + r) * lda + k0 + 2 * c);
            int gn = bn + r;
            rbu[it] = 0u;
            if (gn < N) rbu[it] = *(const unsigned int*)(Bw + (size_t)gn * ldb + k0 + 2 * c);
        }
    };
    auto store_lds = [&](int buf) {
#pragma unroll
        for (int it = 0; it < 8; ++it) {
            int idx = t + it * 256;
            int r = idx >> 4, c = idx & 15;
            sA[buf][r * LDS_PITCH + c] = pack_bf16x2(ra[it].x, ra[it].y);
            sB[buf][r * LDS_PITCH + c] = rbu[it];
        }
    };

    const int ksteps = K >> 5;
    load_global(0);
    store_lds(0);

    for (int ks = 0; ks < ksteps; ++ks) {
        const int cur = ks & 1;
        __syncthreads();                         // LDS buf[cur] ready for everyone
        if (ks + 1 < ksteps) load_global((ks + 1) << 5);   // overlap with WMMA

        bf16x16 af[2], bfm[4];
#pragma unroll
        for (int mi = 0; mi < 2; ++mi)
            af[mi] = load_frag_a(&sA[cur][(wm * 32 + mi * 16 + nl) * LDS_PITCH], half);
#pragma unroll
        for (int ni = 0; ni < 4; ++ni)
            bfm[ni] = load_frag_b(&sB[cur][(wn * 64 + ni * 16 + nl) * LDS_PITCH], half);
#pragma unroll
        for (int mi = 0; mi < 2; ++mi)
#pragma unroll
            for (int ni = 0; ni < 4; ++ni)
                acc[mi][ni] = __builtin_amdgcn_wmma_f32_16x16x32_bf16(
                    false, af[mi], false, bfm[ni], (short)0, acc[mi][ni], false, false);

        if (ks + 1 < ksteps) store_lds(cur ^ 1);
    }

    // ---- epilogue: bias / residual / relu / guarded store ----
#pragma unroll
    for (int mi = 0; mi < 2; ++mi) {
#pragma unroll
        for (int ni = 0; ni < 4; ++ni) {
            int gn = bn + wn * 64 + ni * 16 + nl;
            if (gn < N) {
                float bv = bias ? bias[gn] : 0.f;
#pragma unroll
                for (int r = 0; r < 8; ++r) {
                    long long gm = bm + wm * 32 + mi * 16 + r + 8 * half;
                    float v = acc[mi][ni][r] + bv;
                    if (res)  v += res[(size_t)gm * N + gn];
                    if (relu) v = v > 0.f ? v : 0.f;
                    C[(size_t)gm * N + gn] = v;
                }
            }
        }
    }
}

// ---------------- LayerNorm: one wave per token (256 features) ----------------
__global__ __launch_bounds__(256)
void ln_kernel(const float* __restrict__ x, const float* __restrict__ g,
               const float* __restrict__ b, float* __restrict__ y)
{
    int wv   = (blockIdx.x * blockDim.x + threadIdx.x) >> 5;
    int lane = threadIdx.x & 31;
    const float* xr = x + (size_t)wv * HID;
    float v[8];
    float s = 0.f;
#pragma unroll
    for (int i = 0; i < 8; ++i) { v[i] = xr[lane + i * 32]; s += v[i]; }
#pragma unroll
    for (int off = 16; off >= 1; off >>= 1) s += __shfl_xor(s, off, 32);
    float mean = s * (1.0f / HID);
    float vs = 0.f;
#pragma unroll
    for (int i = 0; i < 8; ++i) { float d = v[i] - mean; vs += d * d; }
#pragma unroll
    for (int off = 16; off >= 1; off >>= 1) vs += __shfl_xor(vs, off, 32);
    float inv = rsqrtf(vs * (1.0f / HID) + 1e-5f);
    float* yr = y + (size_t)wv * HID;
#pragma unroll
    for (int i = 0; i < 8; ++i) {
        int c = lane + i * 32;
        yr[c] = (v[i] - mean) * inv * g[c] + b[c];
    }
}

// ---------------- embedding columns of proj: h += emb[fb] @ proj_w[:,256:266]^T ----------------
__global__ __launch_bounds__(256)
void embadd_kernel(float* __restrict__ h, const int* __restrict__ fb,
                   const float* __restrict__ emb, const float* __restrict__ proj_w)
{
    int tok = blockIdx.x;
    int o = threadIdx.x;
    int cls = fb[tok];
    const float* e = emb + (size_t)cls * EMB_DIM;
    const float* w = proj_w + (size_t)o * (HID + EMB_DIM) + HID;
    float s = 0.f;
#pragma unroll
    for (int i = 0; i < EMB_DIM; ++i) s = fmaf(e[i], w[i], s);
    h[(size_t)tok * HID + o] += s;
}

// ---------------- delta-rule scan ----------------
// One block per batch element b; 16 heads = 16 lane-groups of 16 (8 waves).
// qkvb row (784 f32) for step s+1 is async-copied to LDS while step s computes.
__global__ __launch_bounds__(256)
void scan_kernel(const float* __restrict__ qkvb, float* __restrict__ o)
{
    __shared__ __align__(16) float stok[2][QKVB];

    const int t     = threadIdx.x;
    const int lane  = t & 31;
    const int v     = lane & 15;
    const int gbase = lane & 16;      // shuffle group base within wave
    const int hh    = t >> 4;         // head 0..15
    const int b     = blockIdx.x;     // batch 0..31

    float W[HD];
#pragma unroll
    for (int k = 0; k < HD; ++k) W[k] = 0.f;

    auto stage = [&](int s, int buf) {
        if (t < QKVB / 4) {   // 196 lanes of b128 copies
            const float* g = qkvb + ((size_t)s * BATCH + b) * QKVB + t * 4;
            async_copy_b128(g, &stok[buf][t * 4]);
        }
    };

    stage(0, 0);
    for (int s = 0; s < S_LEN; ++s) {
        const int cur = s & 1;
        if (s + 1 < S_LEN) { stage(s + 1, cur ^ 1); wait_async<1>(); }
        else               { wait_async<0>(); }
        __syncthreads();                      // stok[cur] visible to all waves

        const float* tk = &stok[cur][hh * (3 * HD + 1)];
        float q  = tk[v];
        float kk = tk[HD + v];
        float vv = tk[2 * HD + v];
        float bt = tk[3 * HD];

        // softmax over the 16-lane group (q and k)
        float mq = q, mk = kk;
#pragma unroll
        for (int m = 8; m >= 1; m >>= 1) {
            mq = fmaxf(mq, __shfl_xor(mq, m, 32));
            mk = fmaxf(mk, __shfl_xor(mk, m, 32));
        }
        float eq = __expf(q - mq), ek = __expf(kk - mk);
        float sq = eq, sk = ek;
#pragma unroll
        for (int m = 8; m >= 1; m >>= 1) {
            sq += __shfl_xor(sq, m, 32);
            sk += __shfl_xor(sk, m, 32);
        }
        q  = eq / sq;
        kk = ek / sk;
        bt = 1.0f / (1.0f + __expf(-bt));     // sigmoid

        // v_old[v] = sum_k W[k][v] * k[k]
        float v_old = 0.f;
#pragma unroll
        for (int k = 0; k < HD; ++k) {
            float kb = __shfl(kk, gbase + k, 32);
            v_old = fmaf(W[k], kb, v_old);
        }
        float dv = bt * (vv - v_old);         // v_new - v_old
        // W[k][v] += k[k]*dv[v];  o[v] = sum_k W[k][v]*q[k]  (updated W)
        float ov = 0.f;
#pragma unroll
        for (int k = 0; k < HD; ++k) {
            float kb = __shfl(kk, gbase + k, 32);
            float qb = __shfl(q,  gbase + k, 32);
            W[k] = fmaf(kb, dv, W[k]);
            ov   = fmaf(W[k], qb, ov);
        }
        o[((size_t)s * BATCH + b) * HID + hh * HD + v] = ov;

        __syncthreads();                      // all reads of stok[cur] done before reuse
    }
}

// ---------------- host orchestration ----------------
static void launch_gemm(const float* A, int lda, const __bf16* B, int ldb,
                        const float* bias, const float* res, float* C,
                        int M, int N, int K, int relu, hipStream_t stream)
{
    dim3 grid(M / 128, (N + 127) / 128);
    gemm_wmma_bf16<<<grid, dim3(256), 0, stream>>>(A, lda, B, ldb, bias, res, C, N, K, relu);
}

extern "C" void kernel_launch(void* const* d_in, const int* in_sizes, int n_in,
                              void* d_out, int out_size, void* d_ws, size_t ws_size,
                              hipStream_t stream)
{
    const float* x      = (const float*)d_in[0];
    const int*   fb     = (const int*)  d_in[1];
    const float* fc1_w  = (const float*)d_in[2];
    const float* fc1_b  = (const float*)d_in[3];
    const float* emb    = (const float*)d_in[4];
    const float* proj_w = (const float*)d_in[5];
    const float* proj_b = (const float*)d_in[6];
    const float* ln1_g  = (const float*)d_in[7];
    const float* ln1_b  = (const float*)d_in[8];
    const float* slow_w = (const float*)d_in[9];
    const float* out_w  = (const float*)d_in[10];
    const float* ln2_g  = (const float*)d_in[11];
    const float* ln2_b  = (const float*)d_in[12];
    const float* ff1_w  = (const float*)d_in[13];
    const float* ff1_b  = (const float*)d_in[14];
    const float* ff2_w  = (const float*)d_in[15];
    const float* ff2_b  = (const float*)d_in[16];
    const float* outl_w = (const float*)d_in[17];
    const float* outl_b = (const float*)d_in[18];
    float* out = (float*)d_out;

    char* ws = (char*)d_ws;
    const size_t SZ_H   = (size_t)NTOK * HID * sizeof(float);      // 33.5 MB
    const size_t SZ_BIG = (size_t)NTOK * DFF * sizeof(float);      // 134 MB
    float* buf_h   = (float*)(ws);
    float* buf_ln  = (float*)(ws + SZ_H);
    float* buf_o   = (float*)(ws + 2 * SZ_H);
    float* buf_big = (float*)(ws + 3 * SZ_H);

    // bf16 weight cache (~9 MB), rebuilt deterministically every call
    __bf16* wb      = (__bf16*)(ws + 3 * SZ_H + SZ_BIG);
    __bf16* wb_fc1  = wb;
    __bf16* wb_proj = wb_fc1  + (size_t)HID * IN_DIM;
    __bf16* wb_slow = wb_proj + (size_t)HID * (HID + EMB_DIM);
    __bf16* wb_out  = wb_slow + (size_t)NLAYER * QKVB * HID;
    __bf16* wb_ff1  = wb_out  + (size_t)NLAYER * HID * HID;
    __bf16* wb_ff2  = wb_ff1  + (size_t)NLAYER * DFF * HID;
    __bf16* wb_outl = wb_ff2  + (size_t)NLAYER * HID * DFF;

    auto cvt = [&](const float* s, __bf16* d, size_t n) {
        int n2 = (int)(n / 2);
        cvt_f32_bf16_kernel<<<(n2 + 255) / 256, 256, 0, stream>>>(s, d, n2);
    };
    cvt(fc1_w,  wb_fc1,  (size_t)HID * IN_DIM);
    cvt(proj_w, wb_proj, (size_t)HID * (HID + EMB_DIM));
    cvt(slow_w, wb_slow, (size_t)NLAYER * QKVB * HID);
    cvt(out_w,  wb_out,  (size_t)NLAYER * HID * HID);
    cvt(ff1_w,  wb_ff1,  (size_t)NLAYER * DFF * HID);
    cvt(ff2_w,  wb_ff2,  (size_t)NLAYER * HID * DFF);
    cvt(outl_w, wb_outl, (size_t)NCLS * HID);

    // fc1: t = x @ fc1_w^T + fc1_b    (M=32768, N=256, K=3072)
    launch_gemm(x, IN_DIM, wb_fc1, IN_DIM, fc1_b, nullptr, buf_big,
                NTOK, HID, IN_DIM, 0, stream);
    // proj (first 256 cols): h = t @ proj_w[:, :256]^T + proj_b
    launch_gemm(buf_big, HID, wb_proj, HID + EMB_DIM, proj_b, nullptr, buf_h,
                NTOK, HID, HID, 0, stream);
    // proj (embedding cols): h += emb[fb] @ proj_w[:, 256:266]^T
    embadd_kernel<<<NTOK, 256, 0, stream>>>(buf_h, fb, emb, proj_w);

    for (int l = 0; l < NLAYER; ++l) {
        // --- delta attention ---
        ln_kernel<<<NTOK / 8, 256, 0, stream>>>(buf_h, ln1_g + l * HID, ln1_b + l * HID, buf_ln);
        launch_gemm(buf_ln, HID, wb_slow + (size_t)l * QKVB * HID, HID,
                    nullptr, nullptr, buf_big, NTOK, QKVB, HID, 0, stream);
        scan_kernel<<<BATCH, 256, 0, stream>>>(buf_big, buf_o);
        launch_gemm(buf_o, HID, wb_out + (size_t)l * HID * (NHEAD * HD), NHEAD * HD,
                    nullptr, buf_h, buf_h, NTOK, HID, NHEAD * HD, 0, stream);
        // --- FFN ---
        ln_kernel<<<NTOK / 8, 256, 0, stream>>>(buf_h, ln2_g + l * HID, ln2_b + l * HID, buf_ln);
        launch_gemm(buf_ln, HID, wb_ff1 + (size_t)l * DFF * HID, HID,
                    ff1_b + (size_t)l * DFF, nullptr, buf_big, NTOK, DFF, HID, 1, stream);
        launch_gemm(buf_big, DFF, wb_ff2 + (size_t)l * HID * DFF, DFF,
                    ff2_b + (size_t)l * HID, buf_h, buf_h, NTOK, HID, DFF, 0, stream);
    }

    // classifier: out = h @ outl_w^T + outl_b  (N=1623, guarded tiles)
    launch_gemm(buf_h, HID, wb_outl, HID, outl_b, nullptr, out,
                NTOK, NCLS, HID, 0, stream);
}
